// GNNEncoder_79680233276043
// MI455X (gfx1250) — matrix operand order repaired and checked by reading, not compiled, not run on hardware
//
#include <hip/hip_runtime.h>
#include <math.h>

typedef _Float16 f16;
typedef __attribute__((ext_vector_type(16))) _Float16 v16h;
typedef __attribute__((ext_vector_type(8)))  _Float16 v8h;
typedef __attribute__((ext_vector_type(8)))  float    v8f;
typedef unsigned int u32x4 __attribute__((ext_vector_type(4)));
typedef int          i32x8 __attribute__((ext_vector_type(8)));
typedef int          i32x4 __attribute__((ext_vector_type(4)));

// ---------------------------------------------------------------------------
// Fixed graph (15-bus system), 28 directed edges (symmetric)
// ---------------------------------------------------------------------------
__device__ const int ESRC[28] = {0,1,1,2,2,3,3,4,4,5,2,6,6,7,4,8,8,9,4,10,10,11,8,12,2,13,13,14};
__device__ const int EDST[28] = {1,0,2,1,3,2,4,3,5,4,6,2,7,6,8,4,9,8,10,4,11,10,12,8,13,2,14,13};

// ---------------------------------------------------------------------------
// Workspace layout (byte offsets from d_ws)
// ---------------------------------------------------------------------------
#define WS_AH    0         // 225 f32  : normalized adjacency AHAT
#define WS_B1    4096      // 480 f32  : bias for layer 1 (replicated)
#define WS_B2    8192      // 480 f32  : bias for layer 2 (replicated)
#define WS_BL    12288     // 16  f32  : bias for latent head (padded)
#define WS_G1T   16384     // 480x64  f16 : G1 transposed [N][K], K padded 45->64
#define WS_G3T   81920     // 16x480  f16 : G3 transposed [N][K], N padded 8->16
#define WS_G2T   102400    // 480x480 f16 : G2 transposed [N][K]
// total ~563 KB

// ---------------------------------------------------------------------------
// Kernel 1: build AHAT = D^-1/2 (A+I) D^-1/2  (trivial, 15x15)
// ---------------------------------------------------------------------------
__global__ void build_ahat(float* __restrict__ AH) {
  if (threadIdx.x == 0 && blockIdx.x == 0) {
    float A[15][15];
    for (int i = 0; i < 15; ++i)
      for (int j = 0; j < 15; ++j) A[i][j] = 0.f;
    for (int e = 0; e < 28; ++e) A[ESRC[e]][EDST[e]] = 1.f;
    for (int i = 0; i < 15; ++i) A[i][i] += 1.f;
    float dinv[15];
    for (int i = 0; i < 15; ++i) {
      float s = 0.f;
      for (int j = 0; j < 15; ++j) s += A[i][j];
      dinv[i] = 1.f / sqrtf(s);
    }
    for (int i = 0; i < 15; ++i)
      for (int j = 0; j < 15; ++j)
        AH[i * 15 + j] = dinv[i] * A[i][j] * dinv[j];
  }
}

// ---------------------------------------------------------------------------
// Kernel 2: build fused f16 weight tables (transposed [N][K] for b128 loads)
// G1[k=c*15+j][n=i*32+h] = AHAT[i,j]*W1[c,h]
// G2[k=j*32+g][n=i*32+h] = AHAT[i,j]*W2[g,h]
// G3[k=i*32+h][n=l]      = Wl[h,l]/15      (mean pool folded in)
// ---------------------------------------------------------------------------
__global__ void build_tables(const float* __restrict__ AH,
                             const float* __restrict__ W1, const float* __restrict__ b1,
                             const float* __restrict__ W2, const float* __restrict__ b2,
                             const float* __restrict__ Wl, const float* __restrict__ bl,
                             f16* __restrict__ G1T, f16* __restrict__ G2T, f16* __restrict__ G3T,
                             float* __restrict__ B1v, float* __restrict__ B2v, float* __restrict__ BLv) {
  int idx = blockIdx.x * blockDim.x + threadIdx.x;
  if (idx < 480 * 64) {                       // G1T [n][k], k padded to 64
    int n = idx >> 6, k = idx & 63;
    float v = 0.f;
    if (k < 45) {
      int i = n >> 5, h = n & 31;
      int c = k / 15, j = k - c * 15;
      v = AH[i * 15 + j] * W1[c * 32 + h];
    }
    G1T[idx] = (f16)v;
    return;
  }
  idx -= 480 * 64;
  if (idx < 480 * 480) {                      // G2T [n][k]
    int n = idx / 480, k = idx - n * 480;
    int i = n >> 5, h = n & 31;
    int j = k >> 5, g = k & 31;
    G2T[idx] = (f16)(AH[i * 15 + j] * W2[g * 32 + h]);
    return;
  }
  idx -= 480 * 480;
  if (idx < 16 * 480) {                       // G3T [n][k], n padded to 16
    int n = idx / 480, k = idx - n * 480;
    float v = 0.f;
    if (n < 8) { int h = k & 31; v = Wl[h * 8 + n] * (1.0f / 15.0f); }
    G3T[idx] = (f16)v;
    return;
  }
  idx -= 16 * 480;
  if (idx < 480) { B1v[idx] = b1[idx & 31]; return; }
  idx -= 480;
  if (idx < 480) { B2v[idx] = b2[idx & 31]; return; }
  idx -= 480;
  if (idx < 16)  { BLv[idx] = (idx < 8) ? bl[idx] : 0.f; }
}

// ---------------------------------------------------------------------------
// Kernel 3: zero the recon output (reference returns zeros_like(obs))
// ---------------------------------------------------------------------------
__global__ void zero_recon(float* __restrict__ p, int n) {
  int i = blockIdx.x * blockDim.x + threadIdx.x;
  if (i < n) p[i] = 0.f;
}

// ---------------------------------------------------------------------------
// Main fused kernel. 128 threads = 4 waves; each wave owns 16 batch rows.
// obs tile via TDM (tensor_load_to_lds) -> A-fragments ->
// GEMM1 (obs@G1) -> LDS stage -> h1 A-fragments in VGPRs ->
// GEMM2 (h1@G2, B streamed from L2) -> LDS transpose scratch ->
// GEMM3 (h2@G3 folded into running accumulator) -> tanh*pi -> latent.
// ---------------------------------------------------------------------------
#define WAVES_PER_BLOCK 4
#define STAGE_HALFS 7680   // 16 rows x 480 cols f16 = 15KB per wave

__device__ inline v8f wmma16(v16h a, v16h b, v8f c) {
  return __builtin_amdgcn_wmma_f32_16x16x32_f16(false, a, false, b, (short)0, c,
                                                false, false);
}
// contiguous 16 halfs (B-fragment: lane holds col n, K=bkb..bkb+15)
__device__ inline v16h ld_contig16(const f16* p) {
  union { v16h v; v8h h[2]; } u;
  u.h[0] = *(const v8h*)(p);
  u.h[1] = *(const v8h*)(p + 8);
  return u.v;
}
// A-fragment gather: halfs [p..p+15] (contiguous)
__device__ inline v16h ld_afrag(const f16* p) {
  union { v16h v; v8h h[2]; } u;
  u.h[0] = *(const v8h*)(p);
  u.h[1] = *(const v8h*)(p + 8);
  return u.v;
}

__global__ void __launch_bounds__(128)
gnn_fused(const float* __restrict__ obs,
          const f16* __restrict__ G1T, const f16* __restrict__ G2T,
          const f16* __restrict__ G3T,
          const float* __restrict__ B1v, const float* __restrict__ B2v,
          const float* __restrict__ BLv,
          float* __restrict__ outp, int Btot) {
  __shared__ f16 lds[WAVES_PER_BLOCK * STAGE_HALFS];

  const int lane   = threadIdx.x & 31;
  const int wave   = threadIdx.x >> 5;
  f16* stage       = lds + wave * STAGE_HALFS;
  const int m_lane = lane & 15;        // A: row, B: col, D: col
  const int hi     = lane >> 4;        // lane half
  const int akb    = hi * 16;          // A-frag K base within a 32-chunk
  const int bkb    = hi * 16;          // B-frag K base within a 32-chunk

  const int rowbase = (blockIdx.x * WAVES_PER_BLOCK + wave) * 16;

  // Warm the L2->WGP path for the weight streams while GEMM1 runs.
  {
    const int ncol0 = m_lane;              // first GEMM2 col this lane touches
    __builtin_prefetch(G2T + ncol0 * 480, 0, 1);
    __builtin_prefetch(G2T + (16 + m_lane) * 480, 0, 1);
    __builtin_prefetch(G3T + m_lane * 480, 0, 1);
  }

  // ---- stage the obs tile [16 x 45 f32] into LDS via the Tensor Data Mover ----
  {
    // Tensor DMA descriptor (ISA ch.8): 2D tile, data_size=4B,
    // tensor_dim0=45 (cols), tensor_dim1=remaining rows (HW zero-pads tail),
    // tile = 45 x 16, row stride 45.
    int rows_here = Btot - rowbase;
    if (rows_here > 16) rows_here = 16;
    if (rows_here < 0)  rows_here = 0;
    rows_here = __builtin_amdgcn_readfirstlane(rows_here);
    const unsigned lds_off =
        (unsigned)__builtin_amdgcn_readfirstlane(wave * (STAGE_HALFS * 2));
    unsigned long long ga =
        (unsigned long long)(const void*)(obs + (size_t)rowbase * 45);
    const unsigned ga_lo = __builtin_amdgcn_readfirstlane((unsigned)(ga & 0xFFFFFFFFu));
    const unsigned ga_hi = __builtin_amdgcn_readfirstlane((unsigned)(ga >> 32));

    u32x4 g0;
    g0[0] = 1u;                                    // count=1 (user descriptor)
    g0[1] = lds_off;                               // lds_addr (bytes)
    g0[2] = ga_lo;                                 // global_addr[31:0]
    g0[3] = (ga_hi & 0x01FFFFFFu) | (2u << 30);    // global_addr[56:32] | type=2

    i32x8 g1;
    g1[0] = (2 << 16);                 // workgroup_mask=0, data_size=2 (4B)
    g1[1] = (45 << 16);                // tensor_dim0 = 45 (bits [79:48])
    g1[2] = (rows_here << 16);         // tensor_dim1 = rows (bits [111:80])
    g1[3] = (45 << 16);                // tile_dim0 = 45 (bits [127:112])
    g1[4] = 16;                        // tile_dim1 = 16 rows (bits [143:128])
    g1[5] = 45;                        // tensor_dim0_stride = 45 (bits [207:160])
    g1[6] = 0;
    g1[7] = 0;

    i32x4 g2 = {0, 0, 0, 0};           // unused dims (2D tensor)
    i32x4 g3 = {0, 0, 0, 0};
    i32x8 g4 = {0, 0, 0, 0, 0, 0, 0, 0};  // extra group (clang-23 6-arg form)

    __builtin_amdgcn_tensor_load_to_lds(g0, g1, g2, g3, g4, 0);
    __builtin_amdgcn_s_wait_tensorcnt(0);
  }

  // ---- build obs A-fragments from LDS (K padded 45 -> 64, f32 -> f16) ----
  v16h a_obs[2];
  {
    const float* fs = (const float*)stage;
#pragma unroll
    for (int kc = 0; kc < 2; ++kc) {
      v16h a;
#pragma unroll
      for (int e = 0; e < 16; ++e) {
        const int k = kc * 32 + akb + e;
        float v = (k < 45) ? fs[m_lane * 45 + k] : 0.f;
        a[e] = (f16)v;
      }
      a_obs[kc] = a;
    }
  }

  // ---- GEMM1: h1 = relu(obs @ G1 + B1), stage rows to LDS ----
  for (int nt = 0; nt < 30; ++nt) {
    const int ncol = nt * 16 + m_lane;
    const f16* g1p = G1T + ncol * 64 + bkb;
    v8f acc = {};
    acc = wmma16(a_obs[0], ld_contig16(g1p), acc);
    acc = wmma16(a_obs[1], ld_contig16(g1p + 32), acc);
    const float bias = B1v[ncol];
    const int m0 = hi * 8;
#pragma unroll
    for (int r = 0; r < 8; ++r) {
      float v = fmaxf(acc[r] + bias, 0.f);
      stage[(m0 + r) * 480 + ncol] = (f16)v;
    }
  }
  __syncthreads();

  // ---- lift h1 into A-fragments held in VGPRs (15 K-chunks) ----
  v16h h1f[15];
#pragma unroll
  for (int kc = 0; kc < 15; ++kc) {
    h1f[kc] = ld_afrag(stage + m_lane * 480 + kc * 32 + akb);
  }
  __syncthreads();   // stage region is now free for transpose scratch

  // ---- GEMM2 (h1 @ G2) fused with GEMM3 (h2 @ G3) accumulation ----
  v8f acc3 = {};
  for (int t = 0; t < 15; ++t) {
#pragma unroll
    for (int sub = 0; sub < 2; ++sub) {
      const int nt   = t * 2 + sub;
      const int ncol = nt * 16 + m_lane;
      const f16* g2p = G2T + ncol * 480 + bkb;   // transposed [N][K], L2-resident
      v8f acc = {};
#pragma unroll
      for (int kc = 0; kc < 15; ++kc) {
        acc = wmma16(h1f[kc], ld_contig16(g2p + kc * 32), acc);
      }
      const float bias = B2v[ncol];
      const int c0 = sub * 16 + m_lane;
      const int m0 = hi * 8;
#pragma unroll
      for (int r = 0; r < 8; ++r) {
        float v = fmaxf(acc[r] + bias, 0.f);
        stage[(m0 + r) * 32 + c0] = (f16)v;      // D-layout -> row-major scratch
      }
    }
    __syncthreads();
    // re-read as A-fragment for this 32-wide h2 chunk, fold into GEMM3
    v16h a2 = ld_afrag(stage + m_lane * 32 + akb);
    const f16* g3p = G3T + m_lane * 480 + t * 32 + bkb;
    acc3 = wmma16(a2, ld_contig16(g3p), acc3);
    __syncthreads();
  }

  // ---- epilogue: latent = tanh(acc3 + bl) * pi, cols 0..7 valid ----
  if (m_lane < 8) {
    const float bias = BLv[m_lane];
    const int m0 = hi * 8;
    const size_t latent_base = (size_t)Btot * 45;
#pragma unroll
    for (int r = 0; r < 8; ++r) {
      const int b = rowbase + m0 + r;
      if (b < Btot) {
        float v = tanhf(acc3[r] + bias) * 3.14159265358979323846f;
        outp[latent_base + (size_t)b * 8 + m_lane] = v;
      }
    }
  }
}

// ---------------------------------------------------------------------------
// Host-side launcher (graph-capture safe: only kernel launches on stream)
// ---------------------------------------------------------------------------
extern "C" void kernel_launch(void* const* d_in, const int* in_sizes, int n_in,
                              void* d_out, int out_size, void* d_ws, size_t ws_size,
                              hipStream_t stream) {
  (void)n_in; (void)out_size; (void)ws_size;
  const float* obs = (const float*)d_in[0];
  const float* W1  = (const float*)d_in[1];
  const float* b1  = (const float*)d_in[2];
  const float* W2  = (const float*)d_in[3];
  const float* b2  = (const float*)d_in[4];
  const float* Wl  = (const float*)d_in[5];
  const float* bl  = (const float*)d_in[6];
  float* outp = (float*)d_out;
  const int Btot = in_sizes[0] / 45;

  char* ws = (char*)d_ws;
  float* AH  = (float*)(ws + WS_AH);
  float* B1v = (float*)(ws + WS_B1);
  float* B2v = (float*)(ws + WS_B2);
  float* BLv = (float*)(ws + WS_BL);
  f16* G1T = (f16*)(ws + WS_G1T);
  f16* G3T = (f16*)(ws + WS_G3T);
  f16* G2T = (f16*)(ws + WS_G2T);

  build_ahat<<<1, 32, 0, stream>>>(AH);

  const int tab_elems = 480 * 64 + 480 * 480 + 16 * 480 + 480 + 480 + 16;
  build_tables<<<(tab_elems + 255) / 256, 256, 0, stream>>>(
      AH, W1, b1, W2, b2, Wl, bl, G1T, G2T, G3T, B1v, B2v, BLv);

  const int nrec = Btot * 45;
  zero_recon<<<(nrec + 255) / 256, 256, 0, stream>>>(outp, nrec);

  const int rows_per_block = WAVES_PER_BLOCK * 16;
  gnn_fused<<<(Btot + rows_per_block - 1) / rows_per_block, 128, 0, stream>>>(
      obs, G1T, G2T, G3T, B1v, B2v, BLv, outp, Btot);
}
